// LSTMFromScratch_80204219285727
// MI455X (gfx1250) — compile-verified, matrix operand order checked by reading, not run on hardware
//
#include <hip/hip_runtime.h>
#include <hip/hip_bf16.h>

// ---------------------------------------------------------------------------
// LSTM (fused) for MI455X / gfx1250.
//
// Shapes: B=64, T=2048, IN=256, H=512, OUT=256, gates 4H=2048.
//   k0a/k0b: transpose+convert W_h -> WtH[2048][512] bf16, W_x -> WtX[2048][256]
//   k0c:     convert x -> bf16 (64 MB in ws)
//   k0d:     zero h ping-pong buffers + barrier counters (graph-replay safe)
//   k1:      persistent fused LSTM. 4 groups x 16 WGs (cluster-sized).
//            Group owns 16 batch rows; WG owns 32 hidden cols (128 gate cols).
//            Weight fragments VGPR-resident for the whole sequence (24 x v16bf
//            per wave, uses the CDNA5 1024-VGPR file via s_set_vgpr_msb);
//            h tile staged per step by the Tensor Data Mover (tensor_load_to_lds,
//            TENSORcnt) with TDM pad_enable providing the LDS bank padding;
//            x tile prefetched via global_load_async_to_lds_b128 (ASYNCcnt);
//            gates = bias + x_t@WtX + h@WtH via v_wmma_f32_16x16x32_bf16 on
//            three independent accumulator chains; pointwise fp32; h broadcast
//            via L2 + group barrier (atomic counter + s_cluster_barrier).
//   k2:      y = h_T @ fc_w^T + fc_b via v_wmma_f32_16x16x4_f32.
// ---------------------------------------------------------------------------

#define BATCH 64
#define SEQ   2048
#define INF_  256   // input features
#define HID   512
#define OUTF  256

#define NGROUPS 4
#define WGS_PER_GROUP 16
#define ROWS 16      // batch rows per group
#define SLICE 32     // hidden cols per WG

// LDS pitches (elements); *elemsize multiples of 16 bytes, bank-spread
#define PH  520      // 512 + 8  (TDM pad: 4 DWORDs after each 256-DWORD row)
#define PX  264      // 256 + 8
#define PG  136

// workspace layout (bytes)
#define WS_WTH   0ull
#define WS_WTX   (WS_WTH + 2048ull * 512 * 2)           // 2 MB
#define WS_XBF   (WS_WTX + 2048ull * 256 * 2)           // +1 MB
#define WS_HGLOB (WS_XBF + 64ull * 2048 * 256 * 2)      // +64 MB
#define WS_CTR   (WS_HGLOB + (unsigned long long)NGROUPS * 2 * ROWS * HID * 2)

typedef __attribute__((ext_vector_type(16))) __bf16 v16bf;
typedef __attribute__((ext_vector_type(8)))  float  v8f;
typedef __attribute__((ext_vector_type(2)))  float  v2f;
typedef __attribute__((ext_vector_type(4)))  unsigned v4u;
typedef __attribute__((ext_vector_type(8)))  int      v8i_t;
typedef __attribute__((ext_vector_type(4)))  int      v4i_t;

#define HAVE_TDM __has_builtin(__builtin_amdgcn_tensor_load_to_lds)

__device__ __forceinline__ unsigned short f2bf(float f) {
  unsigned u = __builtin_bit_cast(unsigned, f);
  unsigned r = u + 0x7FFFu + ((u >> 16) & 1u);  // round-to-nearest-even
  return (unsigned short)(r >> 16);
}

__device__ __forceinline__ float sigm(float x)  { return 1.0f / (1.0f + __expf(-x)); }
__device__ __forceinline__ float tanhf_(float x){ return 1.0f - 2.0f / (__expf(2.0f * x) + 1.0f); }

struct U4x2 { uint4 lo, hi; };

// Load one 16x32 bf16 WMMA operand fragment (A or B role).
// p0 points at element [r][k0 + half*8] of a row-major [16][pitch] bf16 array.
__device__ __forceinline__ v16bf ld_bf_frag(const unsigned short* p0) {
  U4x2 t;
  t.lo = *(const uint4*)(p0);        // K +0..7   (lanes 0-15) / +8..15  (lanes 16-31)
  t.hi = *(const uint4*)(p0 + 16);   // K +16..23 (lanes 0-15) / +24..31 (lanes 16-31)
  return __builtin_bit_cast(v16bf, t);
}

// CDNA5 async copy: 16B global -> LDS, tracked by ASYNCcnt.
__device__ __forceinline__ void async_ld16(unsigned lds_off, const void* sbase, unsigned voff) {
  asm volatile("global_load_async_to_lds_b128 %0, %1, %2"
               :: "v"(lds_off), "v"(voff), "s"(sbase) : "memory");
}
template <int N>
__device__ __forceinline__ void async_wait() {
  asm volatile("s_wait_asynccnt %0" :: "n"(N) : "memory");
}

__device__ __forceinline__ unsigned lds_off_of(const void* p) {
  return (unsigned)(unsigned long long)p;  // low 32 bits of generic ptr == LDS byte offset
}

#if HAVE_TDM
// One-shot TDM 2D tile load: 16 rows x tile0 bf16 elements, row stride stride0
// elements, LDS pad appended per row per cfg (D# per CDNA5 ISA 8.3/8.4).
__device__ __forceinline__ void tdm_load_2d(unsigned lds_addr, const void* gaddr,
                                            unsigned cfg, unsigned dim0,
                                            unsigned tile0, unsigned stride0) {
  unsigned long long ga = (unsigned long long)gaddr;
  v4u g0;
  g0[0] = 1u;                                     // count=1 valid user descriptor
  g0[1] = lds_addr;                               // lds_addr (bytes)
  g0[2] = (unsigned)ga;                           // global_addr[31:0]
  g0[3] = (unsigned)(ga >> 32) | 0x80000000u;     // global_addr[56:32] | type=2
  v8i_t g1;
  g1[0] = (int)cfg;                               // mask=0,data_size,pad fields
  g1[1] = (int)((dim0 & 0xFFFFu) << 16);          // tensor_dim0[15:0]
  g1[2] = (int)((dim0 >> 16) | (16u << 16));      // tensor_dim0[31:16] | tensor_dim1=16
  g1[3] = (int)((tile0 & 0xFFFFu) << 16);         // tile_dim0
  g1[4] = 16;                                     // tile_dim1 = 16 rows
  g1[5] = (int)stride0;                           // tensor_dim0_stride[31:0]
  g1[6] = 0;
  g1[7] = 0;
  v4i_t z4 = {0, 0, 0, 0};
  v8i_t z8 = {0, 0, 0, 0, 0, 0, 0, 0};
  // clang-23 / therock signature: (v4u g0, v8i g1, v4i, v4i, v8i, i32 cpol)
  __builtin_amdgcn_tensor_load_to_lds(g0, g1, z4, z4, z8, 0);
}
// cfg: data_size=1(2B)<<16 | pad_enable<<20 | pad_interval<<22 | pad_amount<<25
#define TDM_CFG_H 0x07D10000u   // interval=7 (256 DW = 1KB row), amount=3 (4 DW)
#endif

// --------------------------- k0: preprocessing -----------------------------

// out[N][K] bf16 = transpose(in[K][N] f32)
__global__ void cvt_transpose_kernel(const float* __restrict__ in,
                                     unsigned short* __restrict__ out, int K, int N) {
  int idx = blockIdx.x * 256 + threadIdx.x;
  if (idx >= K * N) return;
  int c = idx / K, k = idx - c * K;
  out[idx] = f2bf(in[k * N + c]);
}

__global__ void cvt_bf16_kernel(const float* __restrict__ in,
                                unsigned short* __restrict__ out, int n) {
  for (int i = blockIdx.x * blockDim.x + threadIdx.x; i < n; i += gridDim.x * blockDim.x)
    out[i] = f2bf(in[i]);
}

__global__ void init_kernel(unsigned* __restrict__ hglob32, unsigned* __restrict__ ctrs) {
  int idx = blockIdx.x * 256 + threadIdx.x;
  if (idx < (int)(NGROUPS * 2 * ROWS * HID * 2 / 4)) hglob32[idx] = 0u;  // h ping-pong = 0
  if (blockIdx.x == 0 && threadIdx.x < 64) ctrs[threadIdx.x] = 0u;
}

// --------------------------- k1: persistent LSTM ---------------------------

__global__ __launch_bounds__(256) void lstm_persistent(
    const float* __restrict__ b4h,            // [2048] fp32 gate bias
    const unsigned short* __restrict__ WtH,   // [2048][512] bf16 (gate-col major)
    const unsigned short* __restrict__ WtX,   // [2048][256] bf16
    const unsigned short* __restrict__ xbf,   // [64][2048][256] bf16
    unsigned short* __restrict__ hglob,       // [NGROUPS][2][16][512] bf16 ping-pong
    unsigned* __restrict__ ctrs,              // [NGROUPS*16] barrier counters
    float* __restrict__ outh, float* __restrict__ outc) {
  __shared__ __attribute__((aligned(16))) unsigned short sH[ROWS * PH];
  __shared__ __attribute__((aligned(16))) unsigned short sX[2][ROWS * PX];
  __shared__ __attribute__((aligned(16))) float sG[ROWS * PG];

  const int tid  = threadIdx.x;
  const int lane = tid & 31;
  const int wid  = tid >> 5;                  // 8 waves, wave owns 16 gate cols
  const int g    = blockIdx.x >> 4;           // batch group 0..3
  const int s    = blockIdx.x & 15;           // hidden slice 0..15

  const int half  = lane >> 4;
  const int half8 = half * 8;
  const int r15   = lane & 15;

  // ---- prologue: weight fragments -> VGPRs (stationary for all 2048 steps) ----
  const int lc = wid * 16 + r15;                       // local gate col 0..127
  const int gc = ((lc >> 5) << 9) + (s << 5) + (lc & 31);  // gate*512 + slice*32 + jj
  const float bv = b4h[gc];
  v16bf bX[8], bH[16];
  {
    const unsigned short* gbx = WtX + gc * INF_ + half8;
    const unsigned short* gbh = WtH + gc * HID + half8;
#pragma unroll
    for (int k = 0; k < 8; ++k)  bX[k] = ld_bf_frag(gbx + k * 32);
#pragma unroll
    for (int k = 0; k < 16; ++k) bH[k] = ld_bf_frag(gbh + k * 32);
  }

  // prefetch x(0) into sX[0] (ASYNCcnt)
  for (int idx = tid; idx < 16 * 32; idx += 256) {
    int r = idx >> 5, ch = idx & 31;
    unsigned xo = (unsigned)((((g * 16 + r) * SEQ) * INF_ + ch * 8) * 2);
    async_ld16(lds_off_of(&sX[0][r * PX + ch * 8]), xbf, xo);
  }

  float cst[2] = {0.f, 0.f};                  // cell state: 2 (row,hidden) pairs/thread
  unsigned* ctr = ctrs + g * 16;              // 64B-spaced counter per group

  const unsigned short* aHp = sH + r15 * PH + half8;

  for (int t = 0; t < SEQ; ++t) {
    // ---- stage h(t) via TDM (TENSORcnt); x rides ASYNCcnt independently ----
    const unsigned hbase = (unsigned)(((g * 2 + (t & 1)) * ROWS) * HID * 2);
#if HAVE_TDM
    if (wid == 0)
      tdm_load_2d(lds_off_of(sH), (const char*)hglob + hbase,
                  TDM_CFG_H, HID, HID, HID);
#else
    for (int idx = tid; idx < 16 * 64; idx += 256) {
      int r = idx >> 6, ch = idx & 63;
      async_ld16(lds_off_of(&sH[r * PH + ch * 8]), hglob,
                 hbase + (unsigned)((r * HID + ch * 8) * 2));
    }
#endif
    async_wait<0>();     // x(t) landed (h is on TENSORcnt, not counted here)
    __syncthreads();     // all lanes' x(t) visible in LDS

    const bool more = (t + 1 < SEQ);
    if (more) {
      for (int idx = tid; idx < 16 * 32; idx += 256) {       // 2 chunks/thread
        int r = idx >> 5, ch = idx & 31;
        unsigned xo = (unsigned)(((((g * 16 + r) * SEQ + t + 1)) * INF_ + ch * 8) * 2);
        async_ld16(lds_off_of(&sX[(t + 1) & 1][r * PX + ch * 8]), xbf, xo);
      }
    }

    // ---- x-part GEMM (K=256) on chain acc0 while h(t) streams in ----
    v8f acc0 = {bv, bv, bv, bv, bv, bv, bv, bv};
    v8f acc1 = {0.f, 0.f, 0.f, 0.f, 0.f, 0.f, 0.f, 0.f};
    v8f acc2 = {0.f, 0.f, 0.f, 0.f, 0.f, 0.f, 0.f, 0.f};
    const unsigned short* aXp = &sX[t & 1][r15 * PX + half8];
#pragma unroll
    for (int k = 0; k < 8; ++k)
      acc0 = __builtin_amdgcn_wmma_f32_16x16x32_bf16(false, ld_bf_frag(aXp + k * 32),
                                                     false, bX[k], (short)0, acc0,
                                                     false, false);

#if HAVE_TDM
    if (wid == 0) __builtin_amdgcn_s_wait_tensorcnt((short)0);  // h(t) landed
#else
    async_wait<2>();  // conservative: x(t+1) (2/thread) may still fly
#endif
    __syncthreads();                                  // all lanes' h(t) visible

    // ---- h-part GEMM (K=512) on two independent chains ----
#pragma unroll
    for (int k = 0; k < 8; ++k)
      acc1 = __builtin_amdgcn_wmma_f32_16x16x32_bf16(false, ld_bf_frag(aHp + k * 32),
                                                     false, bH[k], (short)0, acc1,
                                                     false, false);
#pragma unroll
    for (int k = 8; k < 16; ++k)
      acc2 = __builtin_amdgcn_wmma_f32_16x16x32_bf16(false, ld_bf_frag(aHp + k * 32),
                                                     false, bH[k], (short)0, acc2,
                                                     false, false);

    // D layout: VGPR v -> (m = v + 8*half, n = lane&15)
#pragma unroll
    for (int v = 0; v < 8; ++v)
      sG[(v + half8) * PG + lc] = acc0[v] + acc1[v] + acc2[v];
    __syncthreads();

    // ---- pointwise gate math (fp32), 512 (row,hidden) pairs over 256 thr ----
#pragma unroll
    for (int it = 0; it < 2; ++it) {
      int p = tid + it * 256;
      int r = p >> 5, jj = p & 31;
      float iv = sigm(sG[r * PG + jj]);
      float fv = sigm(sG[r * PG + 32 + jj]);
      float gv = tanhf_(sG[r * PG + 64 + jj]);
      float ov = sigm(sG[r * PG + 96 + jj]);
      float cn = fv * cst[it] + iv * gv;
      cst[it] = cn;
      float hn = ov * tanhf_(cn);
      int hcol = (s << 5) + jj;
      hglob[((g * 2 + ((t + 1) & 1)) * ROWS + r) * HID + hcol] = f2bf(hn);
      if (t == SEQ - 1) {
        int brow = g * 16 + r;
        outh[brow * HID + hcol] = hn;
        outc[brow * HID + hcol] = cn;
      }
    }

    // ---- group barrier: atomic counter (plain launch) + HW cluster barrier ----
    __threadfence();
    __syncthreads();
    if (tid == 0) {
      atomicAdd(ctr, 1u);
      unsigned tgt = (unsigned)(WGS_PER_GROUP * (t + 1));
      while (__hip_atomic_load(ctr, __ATOMIC_RELAXED, __HIP_MEMORY_SCOPE_AGENT) < tgt)
        __builtin_amdgcn_s_sleep(2);
    }
    __builtin_amdgcn_s_cluster_barrier();  // fast path if cluster-launched; NOP otherwise
    __syncthreads();
    __threadfence();
  }
}

// --------------------------- k2: final FC ----------------------------------

__global__ __launch_bounds__(256) void fc_kernel(const float* __restrict__ hT,
                                                 const float* __restrict__ fcw,
                                                 const float* __restrict__ fcb,
                                                 float* __restrict__ y) {
  const int lane = threadIdx.x & 31, wid = threadIdx.x >> 5;
  const int tile = blockIdx.x * 8 + wid;   // 64 tiles: 4 x 16
  const int tm = tile >> 4, tn = tile & 15;
  const int half = lane >> 4, n16 = lane & 15;
  const float bvv = fcb[tn * 16 + n16];
  v8f acc = {bvv, bvv, bvv, bvv, bvv, bvv, bvv, bvv};
  const int arow = tm * 16 + n16;          // A row held by this lane
  const int bcol = tn * 16 + n16;          // B col held by this lane (fc_w row)
#pragma unroll 8
  for (int k0 = 0; k0 < HID; k0 += 4) {
    v2f a = *(const v2f*)(hT + arow * HID + k0 + half * 2);
    v2f b = *(const v2f*)(fcw + bcol * HID + k0 + half * 2);
    acc = __builtin_amdgcn_wmma_f32_16x16x4_f32(false, a, false, b,
                                                (short)0, acc, false, false);
  }
#pragma unroll
  for (int v = 0; v < 8; ++v)
    y[(tm * 16 + v + 8 * half) * OUTF + tn * 16 + n16] = acc[v];
}

// --------------------------- host launch -----------------------------------

extern "C" void kernel_launch(void* const* d_in, const int* in_sizes, int n_in,
                              void* d_out, int out_size, void* d_ws, size_t ws_size,
                              hipStream_t stream) {
  const float* x    = (const float*)d_in[0];
  const float* W_x  = (const float*)d_in[1];
  const float* W_h  = (const float*)d_in[2];
  const float* b    = (const float*)d_in[3];
  const float* fc_w = (const float*)d_in[4];
  const float* fc_b = (const float*)d_in[5];
  float* out = (float*)d_out;  // [y 64x256 | h 64x512 | c 64x512]
  char* ws = (char*)d_ws;

  unsigned short* WtH   = (unsigned short*)(ws + WS_WTH);
  unsigned short* WtX   = (unsigned short*)(ws + WS_WTX);
  unsigned short* xbf   = (unsigned short*)(ws + WS_XBF);
  unsigned short* hglob = (unsigned short*)(ws + WS_HGLOB);
  unsigned*       ctrs  = (unsigned*)(ws + WS_CTR);

  cvt_transpose_kernel<<<(512 * 2048) / 256, 256, 0, stream>>>(W_h, WtH, 512, 2048);
  cvt_transpose_kernel<<<(256 * 2048) / 256, 256, 0, stream>>>(W_x, WtX, 256, 2048);
  cvt_bf16_kernel<<<4096, 256, 0, stream>>>(x, xbf, BATCH * SEQ * INF_);
  init_kernel<<<128, 256, 0, stream>>>((unsigned*)hglob, ctrs);

  // 64 persistent WGs (4 groups x 16); far below MI455X WGP count -> co-resident.
  lstm_persistent<<<NGROUPS * WGS_PER_GROUP, 256, 0, stream>>>(
      b, WtH, WtX, xbf, hglob, ctrs, out + 16384, out + 49152);

  fc_kernel<<<8, 256, 0, stream>>>(out + 16384, fc_w, fc_b, out);
}
